// LoRALayer_44006234915015
// MI455X (gfx1250) — matrix operand order repaired and checked by reading, not compiled, not run on hardware
//
#include <hip/hip_runtime.h>

typedef __attribute__((ext_vector_type(16))) __bf16 v16bf;
typedef __attribute__((ext_vector_type(8)))  __bf16 v8bf;
typedef __attribute__((ext_vector_type(8)))  float  v8f;
typedef __attribute__((ext_vector_type(4)))  float  f4;

namespace {
constexpr int TOKENS = 8192;
constexpr int K_IN   = 4096;
constexpr int N_OUT  = 11008;
constexpr int RANK   = 16;
constexpr float ALPHA_OVER_R = 2.0f;   // 32.0 / 16

constexpr int BM = 128, BN = 256, BK = 32;
constexpr int LDT = 40;                  // padded bf16 stride per LDS tile row
constexpr int KSTEPS = K_IN / BK;        // 128
constexpr int XT_HALVES = BM * LDT;      // one x-tile buffer (5120 halves)
constexpr int WT_HALVES = BN * LDT;      // one W-tile buffer (10240 halves)
constexpr size_t SMEM_BYTES =
    (size_t)(4 * XT_HALVES + 4 * WT_HALVES) * sizeof(__bf16);  // 120 KB

constexpr int NT_M = TOKENS / BM;        // 64 tiles along M
constexpr int NT_N = N_OUT / BN;         // 43 tiles along N
constexpr int GROUP_M = 8;               // L2 reuse group
}

// ---------------------------------------------------------------------------
// Stage 1: T[m, j] = (alpha/r) * sum_k x[m,k] * A[j,k]   (8192 x 16)
// ---------------------------------------------------------------------------
__global__ void lora_stage1(const float* __restrict__ X,
                            const float* __restrict__ A,
                            float* __restrict__ T) {
  const int lane = threadIdx.x & 31;
  const int wave = threadIdx.x >> 5;
  const int m = blockIdx.x * 8 + wave;
  const float* xr = X + (size_t)m * K_IN;

  float acc[RANK];
#pragma unroll
  for (int j = 0; j < RANK; ++j) acc[j] = 0.0f;

  for (int k = lane * 4; k < K_IN; k += 32 * 4) {
    f4 xv = *(const f4*)(xr + k);
#pragma unroll
    for (int j = 0; j < RANK; ++j) {
      f4 av = *(const f4*)(A + (size_t)j * K_IN + k);
      acc[j] = fmaf(xv[0], av[0], acc[j]);
      acc[j] = fmaf(xv[1], av[1], acc[j]);
      acc[j] = fmaf(xv[2], av[2], acc[j]);
      acc[j] = fmaf(xv[3], av[3], acc[j]);
    }
  }
#pragma unroll
  for (int j = 0; j < RANK; ++j) {
#pragma unroll
    for (int off = 16; off > 0; off >>= 1)
      acc[j] += __shfl_xor(acc[j], off, 32);
  }
  if (lane == 0) {
#pragma unroll
    for (int j = 0; j < RANK; ++j)
      T[(size_t)m * RANK + j] = ALPHA_OVER_R * acc[j];
  }
}

// ---------------------------------------------------------------------------
// Stage 2: bf16x3 split-precision GEMM:  out = x @ W^T + bias + T @ B^T
// ---------------------------------------------------------------------------
__device__ __forceinline__ v16bf frag16(const __bf16* p0, const __bf16* p1) {
  v8bf a = *(const v8bf*)p0;
  v8bf b = *(const v8bf*)p1;
  return __builtin_shufflevector(a, b, 0,1,2,3,4,5,6,7,8,9,10,11,12,13,14,15);
}

__device__ __forceinline__ void cvt_store16(const f4 r[4], __bf16* dh, __bf16* dl) {
#pragma unroll
  for (int half = 0; half < 2; ++half) {
    v8bf hv = {};
    v8bf lv = {};
#pragma unroll
    for (int e = 0; e < 8; ++e) {
      float x = r[half * 2 + (e >> 2)][e & 3];
      __bf16 hi = (__bf16)x;                 // RNE to bf16 (high part)
      hv[e] = hi;
      lv[e] = (__bf16)(x - (float)hi);       // residual (low part)
    }
    *(v8bf*)(dh + half * 8) = hv;
    *(v8bf*)(dl + half * 8) = lv;
  }
}

__device__ __forceinline__ float dot4f(f4 a, f4 b) {
  return fmaf(a[0], b[0], fmaf(a[1], b[1], fmaf(a[2], b[2], a[3] * b[3])));
}

__global__ void __launch_bounds__(256, 1)
lora_gemm(const float* __restrict__ X, const float* __restrict__ W,
          const float* __restrict__ bias, const float* __restrict__ Bm,
          const float* __restrict__ T, float* __restrict__ out) {
  extern __shared__ __bf16 smem[];
  __bf16* sXh = smem;                        // [2][XT_HALVES]
  __bf16* sXl = sXh + 2 * XT_HALVES;
  __bf16* sWh = sXl + 2 * XT_HALVES;         // [2][WT_HALVES]
  __bf16* sWl = sWh + 2 * WT_HALVES;

  const int tid    = threadIdx.x;
  const int lane   = tid & 31;
  const int wave   = tid >> 5;
  const int waveM  = wave & 1;               // 2 waves along M (64 rows each)
  const int waveN  = wave >> 1;              // 4 waves along N (64 cols each)
  const int l16    = lane & 15;
  const int hiHalf = lane >> 4;

  // Grouped tile swizzle for L2 reuse: 8 M-tiles per group, M fastest inside.
  const int pid       = blockIdx.x;
  const int per_group = GROUP_M * NT_N;      // 344
  const int group     = pid / per_group;
  const int inb       = pid % per_group;
  const int m0 = (group * GROUP_M + (inb % GROUP_M)) * BM;
  const int n0 = (inb / GROUP_M) * BN;

  // global -> reg staging: x tile 128x32 (1 slice/thread), W tile 256x32 (2)
  const int r  = tid >> 1;                   // 0..127
  const int ch = (tid & 1) << 4;             // 0 or 16
  const float* xg  = X + (size_t)(m0 + r) * K_IN + ch;
  const float* wg0 = W + (size_t)(n0 + r) * K_IN + ch;
  const float* wg1 = W + (size_t)(n0 + r + 128) * K_IN + ch;

  f4 rx[4], rw0[4], rw1[4];
  v8f acc[4][4] = {};

  auto gload = [&](int ks) {
    const f4* xp  = (const f4*)(xg  + (size_t)ks * BK);
    const f4* wp0 = (const f4*)(wg0 + (size_t)ks * BK);
    const f4* wp1 = (const f4*)(wg1 + (size_t)ks * BK);
#pragma unroll
    for (int i = 0; i < 4; ++i) { rx[i] = xp[i]; rw0[i] = wp0[i]; rw1[i] = wp1[i]; }
  };

  auto lstore = [&](int buf) {
    const int ox  = buf * XT_HALVES + r * LDT + ch;
    const int ow0 = buf * WT_HALVES + r * LDT + ch;
    const int ow1 = buf * WT_HALVES + (r + 128) * LDT + ch;
    cvt_store16(rx,  sXh + ox,  sXl + ox);
    cvt_store16(rw0, sWh + ow0, sWl + ow0);
    cvt_store16(rw1, sWh + ow1, sWl + ow1);
  };

  auto compute = [&](int cur) {
    const int cbX = cur * XT_HALVES;
    const int cbW = cur * WT_HALVES;
    const int oA0 = hiHalf ? 8 : 0;          // ISA 16-bit A 16x32 layout
    const int oA1 = hiHalf ? 24 : 16;
    const int oB  = hiHalf * 8;              // ISA 16-bit B 32x16 layout

    v16bf ah[4], al[4];
#pragma unroll
    for (int mt = 0; mt < 4; ++mt) {
      const int row = waveM * 64 + mt * 16 + l16;
      const __bf16* ph = sXh + cbX + row * LDT;
      const __bf16* pl = sXl + cbX + row * LDT;
      ah[mt] = frag16(ph + oA0, ph + oA1);
      al[mt] = frag16(pl + oA0, pl + oA1);
    }
#pragma unroll
    for (int nt = 0; nt < 4; ++nt) {
      const int col = waveN * 64 + nt * 16 + l16;
      const __bf16* ph = sWh + cbW + col * LDT + oB;
      const __bf16* pl = sWl + cbW + col * LDT + oB;
      v16bf bhf = frag16(ph, ph + 16);
      v16bf blf = frag16(pl, pl + 16);
      // Issue order keeps >=4 independent WMMAs between same-acc writes.
#pragma unroll
      for (int mt = 0; mt < 4; ++mt)
        acc[mt][nt] = __builtin_amdgcn_wmma_f32_16x16x32_bf16(
            false, ah[mt], false, bhf, (short)0, acc[mt][nt], false, false);
#pragma unroll
      for (int mt = 0; mt < 4; ++mt)
        acc[mt][nt] = __builtin_amdgcn_wmma_f32_16x16x32_bf16(
            false, al[mt], false, bhf, (short)0, acc[mt][nt], false, false);
#pragma unroll
      for (int mt = 0; mt < 4; ++mt)
        acc[mt][nt] = __builtin_amdgcn_wmma_f32_16x16x32_bf16(
            false, ah[mt], false, blf, (short)0, acc[mt][nt], false, false);
    }
  };

  // Double-buffered pipeline with prefetch two K-steps ahead.
  gload(0);
  lstore(0);
  __syncthreads();
  for (int ks = 0; ks < KSTEPS; ++ks) {
    const int cur = ks & 1;
    if (ks + 1 < KSTEPS) {
      gload(ks + 1);
      if (ks + 2 < KSTEPS) {
        __builtin_prefetch(xg  + (size_t)(ks + 2) * BK, 0, 0);
        __builtin_prefetch(wg0 + (size_t)(ks + 2) * BK, 0, 0);
        __builtin_prefetch(wg1 + (size_t)(ks + 2) * BK, 0, 0);
      }
    }
    compute(cur);
    __syncthreads();
    if (ks + 1 < KSTEPS) {
      lstore(1 - cur);
      __syncthreads();
    }
  }

  // Epilogue: + bias + rank-16 LoRA term (exact f32), then store.
  // C/D layout: VGPR v -> M = v + 8*hiHalf, N = lane%16 (per 16x16 tile).
#pragma unroll
  for (int nt = 0; nt < 4; ++nt) {
    const int n = n0 + waveN * 64 + nt * 16 + l16;
    const float bn = bias[n];
    const f4* Bp = (const f4*)(Bm + (size_t)n * RANK);
    f4 b0 = Bp[0], b1 = Bp[1], b2 = Bp[2], b3 = Bp[3];
#pragma unroll
    for (int mt = 0; mt < 4; ++mt) {
#pragma unroll
      for (int v = 0; v < 8; ++v) {
        const int m = m0 + waveM * 64 + mt * 16 + hiHalf * 8 + v;
        const f4* tp = (const f4*)(T + (size_t)m * RANK);
        float s = bn + dot4f(tp[0], b0) + dot4f(tp[1], b1)
                     + dot4f(tp[2], b2) + dot4f(tp[3], b3);
        out[(size_t)m * N_OUT + n] = acc[mt][nt][v] + s;
      }
    }
  }
}

// ---------------------------------------------------------------------------
extern "C" void kernel_launch(void* const* d_in, const int* in_sizes, int n_in,
                              void* d_out, int out_size, void* d_ws, size_t ws_size,
                              hipStream_t stream) {
  const float* x    = (const float*)d_in[0];   // [8192, 4096]
  const float* W    = (const float*)d_in[1];   // [11008, 4096]
  const float* bias = (const float*)d_in[2];   // [11008]
  const float* B    = (const float*)d_in[3];   // [11008, 16]
  const float* A    = (const float*)d_in[4];   // [16, 4096]
  float* out = (float*)d_out;                  // [8192, 11008]
  float* T   = (float*)d_ws;                   // [8192, 16] scratch (512 KB)

  lora_stage1<<<TOKENS / 8, 256, 0, stream>>>(x, A, T);
  lora_gemm<<<NT_M * NT_N, 256, SMEM_BYTES, stream>>>(x, W, bias, B, T, out);
}
